// DefaultGIN_24721831756435
// MI455X (gfx1250) — matrix operand order repaired and checked by reading, not compiled, not run on hardware
//
#include <hip/hip_runtime.h>
#include <hip/hip_bf16.h>

// ---------- types for WMMA fragments ----------
typedef __attribute__((ext_vector_type(16))) __bf16        v16bf;
typedef __attribute__((ext_vector_type(16))) unsigned short us16;
typedef __attribute__((ext_vector_type(8)))  float          v8f;

#define WPAD 264   // padded LDS row stride (ushorts): 528 B, 16B-aligned, de-conflicted

__device__ __forceinline__ unsigned short f2bf(float f) {
    unsigned int u = __builtin_bit_cast(unsigned int, f);
    unsigned int r = u + 0x7FFFu + ((u >> 16) & 1u);   // round-to-nearest-even
    return (unsigned short)(r >> 16);
}

// ---------- small utility kernels ----------
__global__ void k_zero(float* __restrict__ p, long n) {
    long t = (long)blockIdx.x * blockDim.x + threadIdx.x;
    if (t < n) p[t] = 0.0f;
}

// h[i,:] = z[i,:] = emb[x[i],:]   (D = 256, one float4 per thread)
__global__ void k_embed(const int* __restrict__ x, const float* __restrict__ emb,
                        float* __restrict__ h, float* __restrict__ z, int N) {
    long t = (long)blockIdx.x * blockDim.x + threadIdx.x;
    long total = (long)N * 64;
    if (t >= total) return;
    int i = (int)(t >> 6);
    int c = (int)(t & 63) << 2;
    int row = x[i];
    float4 v = *reinterpret_cast<const float4*>(emb + (size_t)row * 256 + c);
    *reinterpret_cast<float4*>(h + (size_t)i * 256 + c) = v;
    *reinterpret_cast<float4*>(z + (size_t)i * 256 + c) = v;
}

// z[dst,:] += h[src,:]   (64 threads per edge, float4 each, hw f32 atomics)
__global__ void k_edge_agg(const int* __restrict__ src, const int* __restrict__ dst,
                           const float* __restrict__ h, float* __restrict__ z, int E) {
    long t = (long)blockIdx.x * blockDim.x + threadIdx.x;
    long total = (long)E * 64;
    if (t >= total) return;
    int e = (int)(t >> 6);
    int c = (int)(t & 63) << 2;
    int s = src[e];
    int d = dst[e];
    float4 v = *reinterpret_cast<const float4*>(h + (size_t)s * 256 + c);
    float* zp = z + (size_t)d * 256 + c;
    unsafeAtomicAdd(zp + 0, v.x);
    unsafeAtomicAdd(zp + 1, v.y);
    unsafeAtomicAdd(zp + 2, v.z);
    unsafeAtomicAdd(zp + 3, v.w);
}

// sums[batch[i],:] += h[i,:]; counts[batch[i]] += 1
__global__ void k_pool(const float* __restrict__ h, const int* __restrict__ batch,
                       float* __restrict__ sums, float* __restrict__ counts, int N) {
    long t = (long)blockIdx.x * blockDim.x + threadIdx.x;
    long total = (long)N * 64;
    if (t >= total) return;
    int i = (int)(t >> 6);
    int c = (int)(t & 63) << 2;
    int g = batch[i];
    float4 v = *reinterpret_cast<const float4*>(h + (size_t)i * 256 + c);
    float* sp = sums + (size_t)g * 256 + c;
    unsafeAtomicAdd(sp + 0, v.x);
    unsafeAtomicAdd(sp + 1, v.y);
    unsafeAtomicAdd(sp + 2, v.z);
    unsafeAtomicAdd(sp + 3, v.w);
    if (c == 0) unsafeAtomicAdd(counts + g, 1.0f);
}

// out[g,o] = bf[o] + sum_k (sums[g,k]/max(counts[g],1)) * wf[k,o]   (G x 32)
__global__ void k_head(const float* __restrict__ sums, const float* __restrict__ counts,
                       const float* __restrict__ wf, const float* __restrict__ bfv,
                       float* __restrict__ out, int G) {
    int g = blockIdx.x;
    int o = threadIdx.x;           // 32 threads (one wave)
    if (g >= G || o >= 32) return;
    float cnt = counts[g];
    cnt = cnt > 1.0f ? cnt : 1.0f;
    float inv = 1.0f / cnt;
    const float* srow = sums + (size_t)g * 256;
    float acc = bfv[o];
    #pragma unroll 4
    for (int k = 0; k < 256; ++k)
        acc += (srow[k] * inv) * wf[k * 32 + o];
    out[(size_t)g * 32 + o] = acc;
}

// ---------- fragment loaders ----------
// A fragment, 16-bit 16x32 layout: elems 0..7 -> K0+half*8+0..7,
// elems 8..15 -> K0+16+half*8+0..7 (f32 -> bf16 on the fly).
// Branchless: caller passes a clamped (always in-bounds) row pointer.
__device__ __forceinline__ us16 loadA(const float* __restrict__ arow,
                                      int K0, int half) {
    const float4* p = reinterpret_cast<const float4*>(arow + K0 + half * 8);
    float4 f0 = p[0], f1 = p[1];   // K = +0..7
    float4 f2 = p[4], f3 = p[5];   // K = +16..23
    us16 a;
    a[0]  = f2bf(f0.x); a[1]  = f2bf(f0.y); a[2]  = f2bf(f0.z); a[3]  = f2bf(f0.w);
    a[4]  = f2bf(f1.x); a[5]  = f2bf(f1.y); a[6]  = f2bf(f1.z); a[7]  = f2bf(f1.w);
    a[8]  = f2bf(f2.x); a[9]  = f2bf(f2.y); a[10] = f2bf(f2.z); a[11] = f2bf(f2.w);
    a[12] = f2bf(f3.x); a[13] = f2bf(f3.y); a[14] = f2bf(f3.z); a[15] = f2bf(f3.w);
    return a;
}

// B fragment: column n; elems 0..7 -> K0+half*8+0..7, elems 8..15 -> K0+16+half*8+0..7
__device__ __forceinline__ us16 loadB(const unsigned short* __restrict__ Ws,
                                      int n, int K0, int half) {
    const unsigned short* pb = Ws + n * WPAD + K0 + half * 8;
    us16 b;
    #pragma unroll
    for (int j = 0; j < 16; ++j) b[j] = pb[j < 8 ? j : j + 8];
    return b;
}

// ---------- WMMA GEMM:  C[M,256] = act( A[M,256] @ W[256,256] + bias ) ----------
// Block = 256 threads = 8 waves as 4 M-stripes x 2 N-halves (128x256 tile);
// each wave owns 32 rows x 128 cols (2 M-tiles x 8 N-tiles). Per K-step all
// 16 ds_load_b128 for the 8 B fragments are issued, then sched_barrier(0)
// pins them ahead of the 16-WMMA chain, so all 8 fragments stay live and the
// hardware waits become graduated (overlap LDS latency with the matrix pipe).
// Body is branchless (row indices clamped; stores guarded in the epilogue).
__global__ void __launch_bounds__(256)
k_gemm_mlp(const float* __restrict__ A, const float* __restrict__ W,
           const float* __restrict__ bias, float* __restrict__ C,
           float* __restrict__ C2, int M, int do_relu) {
    __shared__ unsigned short Ws[256 * WPAD];   // Ws[n*WPAD + k] = bf16(W[k,n])

    const int tid = threadIdx.x;
    for (int idx = tid; idx < 256 * 256; idx += 256) {
        int k = idx >> 8;
        int n = idx & 255;
        Ws[n * WPAD + k] = f2bf(W[idx]);       // coalesced global read, padded LDS
    }
    __syncthreads();

    const int wave  = tid >> 5;
    const int lane  = tid & 31;
    const int half  = lane >> 4;
    const int lm    = lane & 15;
    const int waveM = wave >> 1;     // 0..3  -> M stripe
    const int waveN = wave & 1;      // 0..1  -> N half

    const int row0  = blockIdx.x * 128 + waveM * 32;   // wave's 32-row stripe
    const int ncol0 = waveN * 128;                     // wave's 128-col stripe
    const int m0 = row0 + lm;
    const int m1 = row0 + 16 + lm;
    const int m0c = m0 < M ? m0 : M - 1;   // clamped: loads always in-bounds,
    const int m1c = m1 < M ? m1 : M - 1;   // bogus rows discarded by store guard
    const float* ar0 = A + (size_t)m0c * 256;
    const float* ar1 = A + (size_t)m1c * 256;

    v8f acc[2][8];
    #pragma unroll
    for (int nt = 0; nt < 8; ++nt) {
        float bv = bias[ncol0 + nt * 16 + lm];
        #pragma unroll
        for (int j = 0; j < 8; ++j) { acc[0][nt][j] = bv; acc[1][nt][j] = bv; }
    }

    for (int kb = 0; kb < 8; ++kb) {
        const int K0 = kb * 32;

        v16bf a0 = __builtin_bit_cast(v16bf, loadA(ar0, K0, half));
        v16bf a1 = __builtin_bit_cast(v16bf, loadA(ar1, K0, half));

        // issue ALL 16 ds_load_b128 for the 8 B fragments ...
        us16 b[8];
        #pragma unroll
        for (int nt = 0; nt < 8; ++nt)
            b[nt] = loadB(Ws, ncol0 + nt * 16 + lm, K0, half);

        // ... and pin them ahead of the WMMA chain
        __builtin_amdgcn_sched_barrier(0);

        #pragma unroll
        for (int nt = 0; nt < 8; ++nt) {
            v16bf bv16 = __builtin_bit_cast(v16bf, b[nt]);
            acc[0][nt] = __builtin_amdgcn_wmma_f32_16x16x32_bf16(
                false, a0, false, bv16, (short)0, acc[0][nt], false, false);
            acc[1][nt] = __builtin_amdgcn_wmma_f32_16x16x32_bf16(
                false, a1, false, bv16, (short)0, acc[1][nt], false, false);
        }
    }

    // epilogue: C/D layout: VGPR j -> row base + half*8 + j, col = ncol0 + nt*16 + lm
    #pragma unroll
    for (int r = 0; r < 2; ++r) {
        const int rbase = row0 + r * 16 + half * 8;
        #pragma unroll
        for (int nt = 0; nt < 8; ++nt) {
            const int col = ncol0 + nt * 16 + lm;
            #pragma unroll
            for (int j = 0; j < 8; ++j) {
                const int row = rbase + j;
                if (row < M) {
                    float v = acc[r][nt][j];
                    if (do_relu) v = fmaxf(v, 0.0f);
                    C[(size_t)row * 256 + col] = v;
                    if (C2) C2[(size_t)row * 256 + col] = v;
                }
            }
        }
    }
}

// ---------- host-side orchestration ----------
extern "C" void kernel_launch(void* const* d_in, const int* in_sizes, int n_in,
                              void* d_out, int out_size, void* d_ws, size_t ws_size,
                              hipStream_t stream) {
    const int*   x    = (const int*)  d_in[0];
    const int*   ei   = (const int*)  d_in[1];   // [2, E]: row 0 = src, row 1 = dst
    const int*   bat  = (const int*)  d_in[2];
    const float* emb  = (const float*)d_in[3];
    const float* w1a  = (const float*)d_in[4];
    const float* b1a  = (const float*)d_in[5];
    const float* w1b  = (const float*)d_in[6];
    const float* b1b  = (const float*)d_in[7];
    const float* w2a  = (const float*)d_in[8];
    const float* b2a  = (const float*)d_in[9];
    const float* w2b  = (const float*)d_in[10];
    const float* b2b  = (const float*)d_in[11];
    const float* wf   = (const float*)d_in[12];
    const float* bfv  = (const float*)d_in[13];
    float*       out  = (float*)d_out;

    const int N = in_sizes[0];
    const int E = in_sizes[1] / 2;
    const int G = out_size / 32;
    const int* src = ei;
    const int* dst = ei + E;

    // workspace layout (f32): h | z | t | sums | counts
    float* hbuf   = (float*)d_ws;
    float* zbuf   = hbuf + (size_t)N * 256;
    float* tbuf   = zbuf + (size_t)N * 256;
    float* sums   = tbuf + (size_t)N * 256;
    float* counts = sums + (size_t)G * 256;

    const int  TB    = 256;
    const long nodeT = (long)N * 64;                    // float4 granularity
    const int  nodeB = (int)((nodeT + TB - 1) / TB);
    const long edgeT = (long)E * 64;
    const int  edgeB = (int)((edgeT + TB - 1) / TB);
    const int  gemmB = (N + 127) / 128;
    const long zlen  = (long)G * 256 + G;
    const int  zB    = (int)((zlen + TB - 1) / TB);

    // zero pooled sums + counts (contiguous)
    k_zero<<<zB, TB, 0, stream>>>(sums, zlen);

    // h0 = emb[x]  (also primes z with the self term)
    k_embed<<<nodeB, TB, 0, stream>>>(x, emb, hbuf, zbuf, N);

    // ---- GIN conv 1 ----
    k_edge_agg<<<edgeB, TB, 0, stream>>>(src, dst, hbuf, zbuf, E);            // z = h + agg
    k_gemm_mlp<<<gemmB, TB, 0, stream>>>(zbuf, w1a, b1a, tbuf, nullptr, N, 1);// t = relu(z@w1a+b1a)
    k_gemm_mlp<<<gemmB, TB, 0, stream>>>(tbuf, w1b, b1b, hbuf, zbuf,   N, 1);// h = z = relu(t@w1b+b1b)

    // ---- GIN conv 2 ----
    k_edge_agg<<<edgeB, TB, 0, stream>>>(src, dst, hbuf, zbuf, E);            // z = h + agg
    k_gemm_mlp<<<gemmB, TB, 0, stream>>>(zbuf, w2a, b2a, tbuf, nullptr, N, 1);
    k_gemm_mlp<<<gemmB, TB, 0, stream>>>(tbuf, w2b, b2b, hbuf, nullptr, N, 0);// no relu

    // ---- global mean pool + head ----
    k_pool<<<nodeB, TB, 0, stream>>>(hbuf, bat, sums, counts, N);
    k_head<<<G, 32, 0, stream>>>(sums, counts, wf, bfv, out, G);
}
